// MultiHeadAttention_33062658244670
// MI455X (gfx1250) — matrix operand order repaired and checked by reading, not compiled, not run on hardware
//
#include <hip/hip_runtime.h>
#include <hip/hip_bf16.h>

// ---------------------------------------------------------------------------
// MHA forward for MI455X (gfx1250, wave32, WMMA bf16 w/ f32 accumulate)
// B=2, S=2048, D_MODEL=1024, H=16, D_K=64
// Fully fused scores+softmax+PV: attn written once (NT), never re-read.
// ---------------------------------------------------------------------------

typedef __attribute__((ext_vector_type(16))) __bf16 v16bf;
typedef __attribute__((ext_vector_type(8)))  float  v8f;

#define BB   2
#define SS   2048
#define HH   16
#define DM   1024
#define DK   64
#define MTOT (BB * SS)                     // 4096 rows of X
#define OUT_ELEMS ((size_t)BB * SS * DM)   // 4,194,304
#define PSTR 40                            // padded LDS row (32 keys + 8 pad)

// ----- fragment loaders ----------------------------------------------------
// A-matrix 16x32 bf16: lane = half*16+m ; per lane k = half*16 + e (16 contig)
//   (matches B-fragment k order)
// caller passes p = rowptr + kchunk + half*16
__device__ __forceinline__ v16bf fragA_f32(const float* __restrict__ p) {
  v16bf f;
#pragma unroll
  for (int i = 0; i < 8; ++i) { f[i] = (__bf16)p[i]; f[i + 8] = (__bf16)p[i + 8]; }
  return f;
}
__device__ __forceinline__ v16bf fragA_bf16(const __bf16* p) {
  v16bf f;
#pragma unroll
  for (int i = 0; i < 8; ++i) { f[i] = p[i]; f[i + 8] = p[i + 8]; }
  return f;
}
// B-matrix 32x16 bf16: lane = half*16+n ; per lane k = half*16 + e (16 contig)
// caller passes p = colptr(k-major) + kchunk + half*16
__device__ __forceinline__ v16bf fragB_f32(const float* __restrict__ p) {
  v16bf f;
#pragma unroll
  for (int i = 0; i < 16; ++i) f[i] = (__bf16)p[i];
  return f;
}
__device__ __forceinline__ v16bf fragB_bf16(const __bf16* __restrict__ p) {
  return *(const v16bf*)p;   // 32B aligned by construction
}

__device__ __forceinline__ v8f wmma_bf16(v16bf a, v16bf b, v8f c) {
  return __builtin_amdgcn_wmma_f32_16x16x32_bf16(false, a, false, b, (short)0, c,
                                                 false, false);
}

// ---------------------------------------------------------------------------
// Kernel 1-3: projection  Y[m,n] = sum_k X[m,k] * W[n,k] + bias[n]  (bf16 out)
// M=4096, N=1024, K=1024.  grid(32,16) block(256): 8 waves, wave = 16x64 tile.
// TRANS_OUT: write Y to [B,1024,S] (for V, so PV B-frags are k-contiguous)
// ---------------------------------------------------------------------------
template <bool TRANS_OUT>
__global__ __launch_bounds__(256) void proj_kernel(const float* __restrict__ X,
                                                   const float* __restrict__ W,
                                                   const float* __restrict__ bias,
                                                   __bf16* __restrict__ Y) {
  const int K = DM, N = DM;
  const int lane = threadIdx.x & 31, wave = threadIdx.x >> 5;
  const int half = lane >> 4, l16 = lane & 15;
  const int m0 = blockIdx.x * 128 + wave * 16;
  const int n0 = blockIdx.y * 64;

  v8f acc[4] = {};
  const float* arow = X + (size_t)(m0 + l16) * K + half * 16;
  const float* brow[4];
#pragma unroll
  for (int t = 0; t < 4; ++t)
    brow[t] = W + (size_t)(n0 + t * 16 + l16) * K + half * 16;

  for (int k0 = 0; k0 < K; k0 += 32) {
    v16bf a = fragA_f32(arow + k0);
#pragma unroll
    for (int t = 0; t < 4; ++t)
      acc[t] = wmma_bf16(a, fragB_f32(brow[t] + k0), acc[t]);
  }
#pragma unroll
  for (int t = 0; t < 4; ++t) {
    const int n = n0 + t * 16 + l16;
    const float bv = bias[n];
#pragma unroll
    for (int r = 0; r < 8; ++r) {
      const int m = m0 + half * 8 + r;
      const float val = acc[t][r] + bv;
      if (TRANS_OUT) {
        const int b = m >> 11, s = m & (SS - 1);
        Y[((size_t)(b * N + n)) * SS + s] = (__bf16)val;
      } else {
        Y[(size_t)m * N + n] = (__bf16)val;
      }
    }
  }
}

// ---------------------------------------------------------------------------
// Kernel 4: FUSED scores + softmax + P.V
// One block owns a 16(q) x 2048(k) strip of one (b,h):
//   phase 1: 8 waves x 16 WMMA tiles -> full score strip in 128 accum VGPRs/wave
//   phase 2: softmax (register + shfl_xor + small LDS reductions)
//   phase 3: store normalized probs to attn (fp32, non-temporal; never re-read)
//   phase 4: each wave transposes ITS OWN 256-key prob slice through a private
//            16x32 LDS buffer (acc layout -> A-frag layout) and accumulates a
//            16x64 partial context over its keys (B = L2-resident Vt)
//   phase 5: cross-wave combine via ds_add_f32, coalesced bf16 Ctx store
// grid(S/16=128, B*H=32) block(256)
// ---------------------------------------------------------------------------
__global__ __launch_bounds__(256) void attn_fused_kernel(
    const __bf16* __restrict__ Qb, const __bf16* __restrict__ Kb,
    const __bf16* __restrict__ Vt, float* __restrict__ attn,
    __bf16* __restrict__ Ctx) {
  const int bh = blockIdx.y, b = bh >> 4, h = bh & 15;
  const int lane = threadIdx.x & 31, wave = threadIdx.x >> 5;
  const int half = lane >> 4, l16 = lane & 15;
  const int q0 = blockIdx.x * 16;
  const int n0 = wave * 256;                 // this wave's 256 key columns

  __shared__ float  red[8][16];
  __shared__ float  fin[16];
  __shared__ __bf16 pbuf[8][16][PSTR];       // per-wave transpose staging
  __shared__ float  pacc[16][DK];            // cross-wave context partials

  // zero context partials (any barrier below orders this vs. the ds_adds)
#pragma unroll
  for (int j = 0; j < 4; ++j)
    pacc[(threadIdx.x * 4 + j) >> 6][(threadIdx.x * 4 + j) & 63] = 0.f;

  // ---- phase 1: scores ----------------------------------------------------
  v8f acc[16] = {};
  const __bf16* arow = Qb + (size_t)(b * SS + q0 + l16) * DM + h * DK + half * 16;
  const __bf16* bbase = Kb + (size_t)(b * SS + n0 + l16) * DM + h * DK + half * 16;

#pragma unroll
  for (int k0 = 0; k0 < DK; k0 += 32) {
    v16bf a = fragA_bf16(arow + k0);
#pragma unroll
    for (int t = 0; t < 16; ++t)
      acc[t] = wmma_bf16(a, fragB_bf16(bbase + (size_t)t * 16 * DM + k0), acc[t]);
  }

  // ---- phase 2: softmax ---------------------------------------------------
  float rmax[8];
#pragma unroll
  for (int r = 0; r < 8; ++r) rmax[r] = -3.402823466e38f;
#pragma unroll
  for (int t = 0; t < 16; ++t)
#pragma unroll
    for (int r = 0; r < 8; ++r) {
      acc[t][r] *= 0.125f;
      rmax[r] = fmaxf(rmax[r], acc[t][r]);
    }
#pragma unroll
  for (int off = 8; off > 0; off >>= 1)
#pragma unroll
    for (int r = 0; r < 8; ++r) rmax[r] = fmaxf(rmax[r], __shfl_xor(rmax[r], off, 32));
  if (l16 == 0) {
#pragma unroll
    for (int r = 0; r < 8; ++r) red[wave][half * 8 + r] = rmax[r];
  }
  __syncthreads();
  if (threadIdx.x < 16) {
    float m = red[0][threadIdx.x];
#pragma unroll
    for (int w = 1; w < 8; ++w) m = fmaxf(m, red[w][threadIdx.x]);
    fin[threadIdx.x] = m;
  }
  __syncthreads();
  float qmax[8];
#pragma unroll
  for (int r = 0; r < 8; ++r) qmax[r] = fin[half * 8 + r];

  float rsum[8];
#pragma unroll
  for (int r = 0; r < 8; ++r) rsum[r] = 0.f;
#pragma unroll
  for (int t = 0; t < 16; ++t)
#pragma unroll
    for (int r = 0; r < 8; ++r) {
      acc[t][r] = __expf(acc[t][r] - qmax[r]);
      rsum[r] += acc[t][r];
    }
#pragma unroll
  for (int off = 8; off > 0; off >>= 1)
#pragma unroll
    for (int r = 0; r < 8; ++r) rsum[r] += __shfl_xor(rsum[r], off, 32);
  __syncthreads();               // everyone done reading fin (maxes)
  if (l16 == 0) {
#pragma unroll
    for (int r = 0; r < 8; ++r) red[wave][half * 8 + r] = rsum[r];
  }
  __syncthreads();
  if (threadIdx.x < 16) {
    float s = red[0][threadIdx.x];
#pragma unroll
    for (int w = 1; w < 8; ++w) s += red[w][threadIdx.x];
    fin[threadIdx.x] = 1.0f / s;
  }
  __syncthreads();
#pragma unroll
  for (int t = 0; t < 16; ++t)
#pragma unroll
    for (int r = 0; r < 8; ++r)
      acc[t][r] *= fin[half * 8 + r];     // normalized probabilities

  // ---- phase 3: stream probs to attn (fp32 NT; attn >> L2, never re-read) -
  float* orow = attn + ((size_t)bh * SS + q0) * SS;
#pragma unroll
  for (int t = 0; t < 16; ++t)
#pragma unroll
    for (int r = 0; r < 8; ++r)
      __builtin_nontemporal_store(acc[t][r],
          &orow[(size_t)(half * 8 + r) * SS + n0 + t * 16 + l16]);

  // ---- phase 4: per-wave partial P.V over own 256-key slice --------------
  v8f pv[4] = {};
  const __bf16* vrow[4];
#pragma unroll
  for (int t = 0; t < 4; ++t)
    vrow[t] = Vt + (size_t)(b * DM + h * DK + t * 16 + l16) * SS + n0 + half * 16;

  for (int c = 0; c < 8; ++c) {            // 8 chunks x 32 keys
    // transpose my acc tiles 2c,2c+1 into my LDS buffer (acc -> A layout)
#pragma unroll
    for (int tt = 0; tt < 2; ++tt)
#pragma unroll
      for (int r = 0; r < 8; ++r)
        pbuf[wave][half * 8 + r][tt * 16 + l16] = (__bf16)acc[2 * c + tt][r];
    // same-wave DS ordering makes the staged data visible to the loads below
    v16bf a = fragA_bf16(&pbuf[wave][l16][half * 16]);
#pragma unroll
    for (int t = 0; t < 4; ++t)
      pv[t] = wmma_bf16(a, fragB_bf16(vrow[t] + c * 32), pv[t]);
  }

  // ---- phase 5: cross-wave combine + Ctx store ---------------------------
#pragma unroll
  for (int t = 0; t < 4; ++t)
#pragma unroll
    for (int r = 0; r < 8; ++r)
      atomicAdd(&pacc[half * 8 + r][t * 16 + l16], pv[t][r]);
  __syncthreads();
  {
    const int q = threadIdx.x >> 4, d4 = (threadIdx.x & 15) * 4;
    __bf16* cp = Ctx + (size_t)(b * SS + q0 + q) * DM + h * DK + d4;
#pragma unroll
    for (int j = 0; j < 4; ++j) cp[j] = (__bf16)pacc[q][d4 + j];
  }
}

// ---------------------------------------------------------------------------
// Kernel 5: out[m,n] = sum_k Ctx[m,k]*Wo[n,k] + bo[n]   (fp32 out)
// ---------------------------------------------------------------------------
__global__ __launch_bounds__(256) void outproj_kernel(const __bf16* __restrict__ Xb,
                                                      const float* __restrict__ W,
                                                      const float* __restrict__ bias,
                                                      float* __restrict__ Y) {
  const int K = DM, N = DM;
  const int lane = threadIdx.x & 31, wave = threadIdx.x >> 5;
  const int half = lane >> 4, l16 = lane & 15;
  const int m0 = blockIdx.x * 128 + wave * 16;
  const int n0 = blockIdx.y * 64;

  v8f acc[4] = {};
  const __bf16* arow = Xb + (size_t)(m0 + l16) * K + half * 16;
  const float* brow[4];
#pragma unroll
  for (int t = 0; t < 4; ++t)
    brow[t] = W + (size_t)(n0 + t * 16 + l16) * K + half * 16;

  for (int k0 = 0; k0 < K; k0 += 32) {
    v16bf a = fragA_bf16(arow + k0);
#pragma unroll
    for (int t = 0; t < 4; ++t)
      acc[t] = wmma_bf16(a, fragB_f32(brow[t] + k0), acc[t]);
  }
#pragma unroll
  for (int t = 0; t < 4; ++t) {
    const int n = n0 + t * 16 + l16;
    const float bv = bias[n];
#pragma unroll
    for (int r = 0; r < 8; ++r) {
      const int m = m0 + half * 8 + r;
      Y[(size_t)m * N + n] = acc[t][r] + bv;
    }
  }
}

// ---------------------------------------------------------------------------
extern "C" void kernel_launch(void* const* d_in, const int* in_sizes, int n_in,
                              void* d_out, int out_size, void* d_ws, size_t ws_size,
                              hipStream_t stream) {
  (void)in_sizes; (void)n_in; (void)out_size; (void)ws_size;
  const float* query = (const float*)d_in[0];
  const float* key_  = (const float*)d_in[1];
  const float* value = (const float*)d_in[2];
  const float* Wq = (const float*)d_in[3];
  const float* bq = (const float*)d_in[4];
  const float* Wk = (const float*)d_in[5];
  const float* bk = (const float*)d_in[6];
  const float* Wv = (const float*)d_in[7];
  const float* bv = (const float*)d_in[8];
  const float* Wo = (const float*)d_in[9];
  const float* bo = (const float*)d_in[10];

  float* out  = (float*)d_out;
  float* attn = out + OUT_ELEMS;          // (B,H,S,S) fp32

  __bf16* Qb  = (__bf16*)d_ws;            // [B*S, 1024]
  __bf16* Kb  = Qb + OUT_ELEMS;           // [B*S, 1024]
  __bf16* Vt  = Kb + OUT_ELEMS;           // [B, 1024, S]  (transposed V)
  __bf16* Ctx = Vt + OUT_ELEMS;           // [B*S, 1024]   -> 32 MB total

  const dim3 blk(256);
  proj_kernel<false><<<dim3(MTOT / 128, DM / 64), blk, 0, stream>>>(query, Wq, bq, Qb);
  proj_kernel<false><<<dim3(MTOT / 128, DM / 64), blk, 0, stream>>>(key_,  Wk, bk, Kb);
  proj_kernel<true ><<<dim3(MTOT / 128, DM / 64), blk, 0, stream>>>(value, Wv, bv, Vt);
  attn_fused_kernel<<<dim3(SS / 16, BB * HH), blk, 0, stream>>>(Qb, Kb, Vt, attn, Ctx);
  outproj_kernel<<<dim3(MTOT / 128, DM / 64), blk, 0, stream>>>(Ctx, Wo, bo, out);
}
